// TransformerEncoderLayer_3255585210767
// MI455X (gfx1250) — matrix operand order repaired and checked by reading, not compiled, not run on hardware
//
#include <hip/hip_runtime.h>

typedef unsigned int u32;
typedef __attribute__((ext_vector_type(4)))  u32    u32x4;
typedef __attribute__((ext_vector_type(4)))  int    i32x4;
typedef __attribute__((ext_vector_type(16))) __bf16 v16bf;
typedef __attribute__((ext_vector_type(8)))  float  v8f;

#define B_   4
#define T_   2048
#define D_   1024
#define H_   16
#define DK_  64
#define DFF_ 4096
#define NTOK (B_ * T_)      // 8192
#define MAXD 128
#define NBUCK 255

#if defined(__gfx1250__) && __has_builtin(__builtin_amdgcn_global_load_async_to_lds_b128)
#define HAVE_ASYNC 1
typedef __attribute__((address_space(1))) i32x4 as1_i32x4;
typedef __attribute__((address_space(3))) i32x4 as3_i32x4;
#define ASYNC_B128(gsrc, ldst) \
  __builtin_amdgcn_global_load_async_to_lds_b128((as1_i32x4*)(gsrc), (as3_i32x4*)(ldst), 0, 0)
#else
#define HAVE_ASYNC 0
#endif

union Frag16 { u32x4 u[2]; v16bf v; };

__device__ __forceinline__ __bf16 f2bf(float f) {
  union { float f; u32 u; } a; a.f = f;
  u32 r = (a.u + 0x7FFFu + ((a.u >> 16) & 1u)) >> 16;
  union { unsigned short s; __bf16 b; } o; o.s = (unsigned short)r;
  return o.b;
}

__device__ __forceinline__ void wait_async0() {
#if HAVE_ASYNC
#if __has_builtin(__builtin_amdgcn_s_wait_asynccnt)
  __builtin_amdgcn_s_wait_asynccnt(0);
#else
  asm volatile("s_wait_asynccnt 0x0" ::: "memory");
#endif
#endif
}

// ---------------------------------------------------------------------------
// Generic bf16 WMMA GEMM:  C[M,N] = A[M,K] @ B^T[N,K] + bias  (B stored N-major)
// Block: 256 threads = 8 waves; tile 128(M) x 128(N); wave tile 64 x 32.
// Double-buffered LDS; tiles streamed with GLOBAL_LOAD_ASYNC_TO_LDS (ASYNCcnt)
// so the async DMA of tile i+1 overlaps the 8 WMMAs of tile i.
// ---------------------------------------------------------------------------
template <bool RELU, bool OUT_BF16>
__global__ __launch_bounds__(256) void gemm_wmma(
    const __bf16* __restrict__ A, const __bf16* __restrict__ Bm,
    const float* __restrict__ bias, float* __restrict__ Cf,
    __bf16* __restrict__ Cb, int M, int N, int K)
{
  __shared__ __attribute__((aligned(16))) __bf16 sA[2][128][32];
  __shared__ __attribute__((aligned(16))) __bf16 sB[2][128][32];

  const int t     = threadIdx.x;
  const int lane  = t & 31;
  const int wave  = t >> 5;
  const int lmod  = lane & 15;
  const int lhalf = lane >> 4;
  const int wm    = wave & 1;        // wave M index (0..1)  -> 64 rows
  const int wn    = wave >> 1;       // wave N index (0..3)  -> 32 cols
  const int m0    = blockIdx.y * 128;
  const int n0    = blockIdx.x * 128;
  const int lrow  = t >> 1;          // cooperative loader: row 0..127
  const int lseg  = t & 1;           // 32-byte half of the 64B row

  auto issueTile = [&](int k0, int buf) {
    const __bf16* ga = A  + (size_t)(m0 + lrow) * K + k0 + lseg * 16;
    const __bf16* gb = Bm + (size_t)(n0 + lrow) * K + k0 + lseg * 16;
#if HAVE_ASYNC
    ASYNC_B128(ga,     &sA[buf][lrow][lseg * 16]);
    ASYNC_B128(ga + 8, &sA[buf][lrow][lseg * 16 + 8]);
    ASYNC_B128(gb,     &sB[buf][lrow][lseg * 16]);
    ASYNC_B128(gb + 8, &sB[buf][lrow][lseg * 16 + 8]);
#else
    *(u32x4*)&sA[buf][lrow][lseg * 16]     = *(const u32x4*)ga;
    *(u32x4*)&sA[buf][lrow][lseg * 16 + 8] = *(const u32x4*)(ga + 8);
    *(u32x4*)&sB[buf][lrow][lseg * 16]     = *(const u32x4*)gb;
    *(u32x4*)&sB[buf][lrow][lseg * 16 + 8] = *(const u32x4*)(gb + 8);
#endif
  };

  v8f acc[4][2];
#pragma unroll
  for (int i = 0; i < 4; ++i)
#pragma unroll
    for (int j = 0; j < 2; ++j) acc[i][j] = 0.f;

  const int nk = K >> 5;             // K / 32 tiles
  issueTile(0, 0);

  for (int it = 0; it < nk; ++it) {
    const int buf = it & 1;
    wait_async0();                   // tile `it` landed in LDS (this wave's DMA)
    __syncthreads();                 // everyone's DMA done + prev reads of buf^1 done
    if (it + 1 < nk) issueTile((it + 1) << 5, buf ^ 1);   // overlap with WMMAs

    Frag16 fb[2];
#pragma unroll
    for (int j = 0; j < 2; ++j) {          // B frag: lane = col, K across regs
      const __bf16* pb = &sB[buf][wn * 32 + j * 16 + lmod][lhalf * 8];
      fb[j].u[0] = *(const u32x4*)pb;
      fb[j].u[1] = *(const u32x4*)(pb + 16);
    }
#pragma unroll
    for (int i = 0; i < 4; ++i) {          // A frag: lane = row, K split low/hi half
      Frag16 fa;
      const __bf16* pa = &sA[buf][wm * 64 + i * 16 + lmod][lhalf * 8];
      fa.u[0] = *(const u32x4*)pa;
      fa.u[1] = *(const u32x4*)(pa + 16);
#pragma unroll
      for (int j = 0; j < 2; ++j)
        acc[i][j] = __builtin_amdgcn_wmma_f32_16x16x32_bf16(
            false, fa.v, false, fb[j].v, (short)0, acc[i][j], false, false);
    }
  }

#pragma unroll
  for (int i = 0; i < 4; ++i)
#pragma unroll
    for (int j = 0; j < 2; ++j) {
      const int col = n0 + wn * 32 + j * 16 + lmod;
      const float bv = bias ? bias[col] : 0.f;
#pragma unroll
      for (int r = 0; r < 8; ++r) {
        const int row = m0 + wm * 64 + i * 16 + r + 8 * lhalf;
        float v = acc[i][j][r] + bv;
        if (RELU) v = fmaxf(v, 0.f);
        const size_t idx = (size_t)row * N + col;
        if constexpr (OUT_BF16) Cb[idx] = f2bf(v);
        else                    Cf[idx] = v;
      }
    }
}

// ---------------------------------------------------------------------------
// One-shot per-head V transpose: qkv [tok][3D] (V block) -> vt [B*H][DK][T].
// LDS-tiled so both global read and write are coalesced 32B accesses.
// After this, attention's V-tile fill is a pure async B128 stream.
// ---------------------------------------------------------------------------
__global__ __launch_bounds__(256) void vtrans_kernel(
    const __bf16* __restrict__ qkv, __bf16* __restrict__ vt)
{
  __shared__ __attribute__((aligned(16))) __bf16 tile[64][72];  // 64 tok x 64 dk (+pad)
  const int t0  = blockIdx.x * 64;
  const int bh  = blockIdx.y;
  const int b   = bh >> 4, h = bh & 15;
  const int tr  = threadIdx.x >> 2;     // 0..63
  const int seg = threadIdx.x & 3;      // 16-element (32B) segment

  {
    const __bf16* src = qkv + (size_t)(b * T_ + t0 + tr) * (3 * D_) + 2 * D_ + h * DK_ + seg * 16;
    *(u32x4*)&tile[tr][seg * 16]     = *(const u32x4*)src;
    *(u32x4*)&tile[tr][seg * 16 + 8] = *(const u32x4*)(src + 8);
  }
  __syncthreads();

  const int dk = tr;
  __attribute__((aligned(16))) __bf16 tmp[16];
#pragma unroll
  for (int e = 0; e < 16; ++e) tmp[e] = tile[seg * 16 + e][dk];
  __bf16* dst = vt + ((size_t)bh * DK_ + dk) * T_ + t0 + seg * 16;
  *(u32x4*)dst       = *(u32x4*)&tmp[0];
  *(u32x4*)(dst + 8) = *(u32x4*)&tmp[8];
}

// ---------------------------------------------------------------------------
// Flash attention with T5 relative-position bias.
// grid = (T/128, B*H); 8 waves; each wave owns 16 query rows.
// qkv layout: [tok][3*D], Q at col 0, K at col D; V comes pre-transposed from
// vt [B*H][DK][T], so both K and V tiles stream via GLOBAL_LOAD_ASYNC_TO_LDS.
// Bias bucket saturates outside |key-q| < 128, so tiles entirely outside the
// window take a wave-uniform constant-bias fast path (no per-element clamps).
// ---------------------------------------------------------------------------
__global__ __launch_bounds__(256) void attn_wmma(
    const __bf16* __restrict__ qkv, const __bf16* __restrict__ vt,
    const float* __restrict__ bias_table, __bf16* __restrict__ ctx)
{
  __shared__ __attribute__((aligned(16))) __bf16 sK[32][64];      // [key][dk]
  __shared__ __attribute__((aligned(16))) __bf16 sVt[64][32];     // [dk][key]
  __shared__ __attribute__((aligned(16))) __bf16 sP[8][16][32];   // per-wave P
  __shared__ float sBias[256];

  const int t     = threadIdx.x;
  const int lane  = t & 31;
  const int wave  = t >> 5;
  const int lmod  = lane & 15;
  const int lhalf = lane >> 4;
  const int bh    = blockIdx.y;
  const int b     = bh >> 4;
  const int h     = bh & 15;
  const int q0    = blockIdx.x * 128 + wave * 16;

  if (t < NBUCK) sBias[t] = bias_table[t * H_ + h];
  if (t == NBUCK) sBias[NBUCK] = 0.f;

  // Persistent Q fragments (A-frag, 16 rows x 64 dk -> 2 frags of K=32)
  Frag16 fq[2];
  {
    const __bf16* qp = qkv + (size_t)(b * T_ + q0 + lmod) * (3 * D_) + h * DK_;
#pragma unroll
    for (int d = 0; d < 2; ++d) {
      const __bf16* p = qp + d * 32 + lhalf * 8;
      fq[d].u[0] = *(const u32x4*)p;
      fq[d].u[1] = *(const u32x4*)(p + 16);
    }
  }

  float mrow[8], lrow[8];
  v8f o[4];
#pragma unroll
  for (int r = 0; r < 8; ++r) { mrow[r] = -1e30f; lrow[r] = 0.f; }
#pragma unroll
  for (int j = 0; j < 4; ++j) o[j] = 0.f;

  const int key_t = t >> 3;   // K loader: key row 0..31, 8-elem chunk
  const int chunk = t & 7;
  const int vdk   = t >> 2;   // V loader: dk row 0..63, 8-key segment
  const int vseg  = t & 3;

  for (int kk0 = 0; kk0 < T_; kk0 += 32) {
    __syncthreads();   // previous iteration's readers done before refill
    {
      const __bf16* ks = qkv + (size_t)(b * T_ + kk0 + key_t) * (3 * D_) + D_ + h * DK_ + chunk * 8;
      const __bf16* vs = vt + ((size_t)bh * DK_ + vdk) * T_ + kk0 + vseg * 8;
#if HAVE_ASYNC
      ASYNC_B128(ks, &sK[key_t][chunk * 8]);
      ASYNC_B128(vs, &sVt[vdk][vseg * 8]);
#else
      *(u32x4*)&sK[key_t][chunk * 8] = *(const u32x4*)ks;
      *(u32x4*)&sVt[vdk][vseg * 8]   = *(const u32x4*)vs;
#endif
    }
    wait_async0();
    __syncthreads();

    // ---- scores: S = (Q @ K^T) / 8 + bias -------------------------------
    float s[2][8];
    const int dlo = kk0 - (q0 + 15);        // min over tile of (key - q)
    const int dhi = (kk0 + 31) - q0;        // max over tile of (key - q)
#pragma unroll
    for (int j = 0; j < 2; ++j) {
      Frag16 fk[2];
#pragma unroll
      for (int d = 0; d < 2; ++d) {   // B frag from sK (row=key, K=dk)
        const __bf16* pk = &sK[j * 16 + lmod][d * 32 + lhalf * 8];
        fk[d].u[0] = *(const u32x4*)pk;
        fk[d].u[1] = *(const u32x4*)(pk + 16);
      }
      v8f accs = 0.f;
      accs = __builtin_amdgcn_wmma_f32_16x16x32_bf16(
          false, fq[0].v, false, fk[0].v, (short)0, accs, false, false);
      accs = __builtin_amdgcn_wmma_f32_16x16x32_bf16(
          false, fq[1].v, false, fk[1].v, (short)0, accs, false, false);
      if (dlo >= MAXD - 1) {               // whole tile beyond +window
        const float c = sBias[NBUCK - 1];
#pragma unroll
        for (int r = 0; r < 8; ++r) s[j][r] = fmaf(accs[r], 0.125f, c);
      } else if (dhi <= -(MAXD - 1)) {     // whole tile beyond -window
        const float c = sBias[0];
#pragma unroll
        for (int r = 0; r < 8; ++r) s[j][r] = fmaf(accs[r], 0.125f, c);
      } else {                             // near-diagonal tile: exact buckets
        const int key = kk0 + j * 16 + lmod;
#pragma unroll
        for (int r = 0; r < 8; ++r) {
          const int q = q0 + r + 8 * lhalf;
          int rel = key - q + (MAXD - 1);
          rel = rel < 0 ? 0 : (rel > NBUCK - 1 ? NBUCK - 1 : rel);
          s[j][r] = fmaf(accs[r], 0.125f, sBias[rel]);
        }
      }
    }

    // ---- online softmax (row lives in a 16-lane half-wave) --------------
#pragma unroll
    for (int r = 0; r < 8; ++r) {
      float mx = fmaxf(s[0][r], s[1][r]);
#pragma unroll
      for (int off = 1; off < 16; off <<= 1) mx = fmaxf(mx, __shfl_xor(mx, off, 32));
      const float mn    = fmaxf(mrow[r], mx);
      const float alpha = __expf(mrow[r] - mn);
      mrow[r] = mn;
      const float p0 = __expf(s[0][r] - mn);
      const float p1 = __expf(s[1][r] - mn);
      float rs = p0 + p1;
#pragma unroll
      for (int off = 1; off < 16; off <<= 1) rs += __shfl_xor(rs, off, 32);
      lrow[r] = lrow[r] * alpha + rs;
#pragma unroll
      for (int j = 0; j < 4; ++j) o[j][r] *= alpha;
      sP[wave][r + 8 * lhalf][lmod]      = f2bf(p0);   // C-layout -> LDS
      sP[wave][r + 8 * lhalf][16 + lmod] = f2bf(p1);
    }
    asm volatile("s_wait_dscnt 0x0" ::: "memory");     // per-wave LDS RAW

    // ---- O += P @ V ------------------------------------------------------
    Frag16 fp;                                         // A-frag of P (16x32)
    const __bf16* pp = &sP[wave][lmod][lhalf * 8];
    fp.u[0] = *(const u32x4*)pp;
    fp.u[1] = *(const u32x4*)(pp + 16);
#pragma unroll
    for (int j = 0; j < 4; ++j) {                      // B-frag from sVt
      Frag16 fv;
      const __bf16* pv = &sVt[j * 16 + lmod][lhalf * 8];
      fv.u[0] = *(const u32x4*)pv;
      fv.u[1] = *(const u32x4*)(pv + 16);
      o[j] = __builtin_amdgcn_wmma_f32_16x16x32_bf16(
          false, fp.v, false, fv.v, (short)0, o[j], false, false);
    }
  }

#pragma unroll
  for (int j = 0; j < 4; ++j)
#pragma unroll
    for (int r = 0; r < 8; ++r) {
      const float v = o[j][r] / lrow[r];
      const int q = q0 + r + 8 * lhalf;
      ctx[(size_t)(b * T_ + q) * D_ + h * DK_ + j * 16 + lmod] = f2bf(v);
    }
}

// ---------------------------------------------------------------------------
// Fused residual + LayerNorm. One block per row of D=1024; 256 threads x 4.
// ---------------------------------------------------------------------------
__global__ __launch_bounds__(256) void add_ln(
    const float* __restrict__ xa, const float* __restrict__ xb,
    const float* __restrict__ g, const float* __restrict__ be,
    float* __restrict__ outf, __bf16* __restrict__ outb)
{
  __shared__ float r1[8], r2[8];
  const int row  = blockIdx.x;
  const int t    = threadIdx.x;
  const int lane = t & 31;
  const int wave = t >> 5;
  const size_t base = (size_t)row * D_;

  float v[4];
  float s1 = 0.f, s2 = 0.f;
#pragma unroll
  for (int i = 0; i < 4; ++i) {
    const int c = t + i * 256;
    v[i] = xa[base + c] + xb[base + c];
    s1 += v[i];
    s2 += v[i] * v[i];
  }
#pragma unroll
  for (int off = 1; off < 32; off <<= 1) {
    s1 += __shfl_xor(s1, off, 32);
    s2 += __shfl_xor(s2, off, 32);
  }
  if (lane == 0) { r1[wave] = s1; r2[wave] = s2; }
  __syncthreads();
  s1 = 0.f; s2 = 0.f;
#pragma unroll
  for (int w = 0; w < 8; ++w) { s1 += r1[w]; s2 += r2[w]; }
  const float mean = s1 * (1.f / D_);
  const float var  = s2 * (1.f / D_) - mean * mean;
  const float rs   = rsqrtf(var + 1e-5f);
#pragma unroll
  for (int i = 0; i < 4; ++i) {
    const int c = t + i * 256;
    const float ov = (v[i] - mean) * rs * g[c] + be[c];
    outf[base + c] = ov;
    if (outb) outb[base + c] = f2bf(ov);
  }
}

// ---------------------------------------------------------------------------
// Small prep kernels.
// ---------------------------------------------------------------------------
__global__ void cast_bf16_kernel(const float* __restrict__ s, __bf16* __restrict__ d, int n) {
  const int i = blockIdx.x * 256 + threadIdx.x;
  if (i < n) d[i] = f2bf(s[i]);
}

// src [K][N] (row-major) -> dst [N][K] bf16 (weights pre-transposed for B-frags)
__global__ void transpose_cast_kernel(const float* __restrict__ s, __bf16* __restrict__ d,
                                      int K, int N) {
  const int i = blockIdx.x * 256 + threadIdx.x;
  if (i < K * N) {
    const int n = i / K, k = i % K;
    d[i] = f2bf(s[(size_t)k * N + n]);
  }
}

__global__ void concat_bias_kernel(const float* __restrict__ bq, const float* __restrict__ bk,
                                   const float* __restrict__ bv, float* __restrict__ o) {
  const int i = blockIdx.x * 256 + threadIdx.x;
  if (i < 3 * D_) o[i] = i < D_ ? bq[i] : (i < 2 * D_ ? bk[i - D_] : bv[i - 2 * D_]);
}

// ---------------------------------------------------------------------------
extern "C" void kernel_launch(void* const* d_in, const int* in_sizes, int n_in,
                              void* d_out, int out_size, void* d_ws, size_t ws_size,
                              hipStream_t stream)
{
  const float* x          = (const float*)d_in[0];
  const float* Wq         = (const float*)d_in[1];
  const float* bq         = (const float*)d_in[2];
  const float* Wk         = (const float*)d_in[3];
  const float* bk         = (const float*)d_in[4];
  const float* Wv         = (const float*)d_in[5];
  const float* bv         = (const float*)d_in[6];
  const float* Wo         = (const float*)d_in[7];
  const float* bo         = (const float*)d_in[8];
  const float* bias_table = (const float*)d_in[9];
  const float* ln1_g      = (const float*)d_in[10];
  const float* ln1_b      = (const float*)d_in[11];
  const float* W1         = (const float*)d_in[12];
  const float* b1         = (const float*)d_in[13];
  const float* W2         = (const float*)d_in[14];
  const float* b2         = (const float*)d_in[15];
  const float* ln2_g      = (const float*)d_in[16];
  const float* ln2_b      = (const float*)d_in[17];
  float* out = (float*)d_out;

  char* w = (char*)d_ws;
  auto alloc = [&](size_t bytes) {
    char* p = w;
    w += (bytes + 255) & ~(size_t)255;
    return p;
  };
  __bf16* xb    = (__bf16*)alloc((size_t)NTOK * D_ * 2);
  __bf16* wqkvT = (__bf16*)alloc((size_t)3 * D_ * D_ * 2);
  __bf16* woT   = (__bf16*)alloc((size_t)D_ * D_ * 2);
  __bf16* w1T   = (__bf16*)alloc((size_t)DFF_ * D_ * 2);
  __bf16* w2T   = (__bf16*)alloc((size_t)D_ * DFF_ * 2);
  float*  bqkv  = (float*) alloc((size_t)3 * D_ * 4);
  __bf16* qkv   = (__bf16*)alloc((size_t)NTOK * 3 * D_ * 2);
  __bf16* vt    = (__bf16*)alloc((size_t)NTOK * D_ * 2);   // V^T per head
  __bf16* ctxb  = (__bf16*)alloc((size_t)NTOK * D_ * 2);
  float*  attnf = (float*) alloc((size_t)NTOK * D_ * 4);   // reused as ff2
  float*  x1f   = (float*) alloc((size_t)NTOK * D_ * 4);
  __bf16* x1b   = (__bf16*)alloc((size_t)NTOK * D_ * 2);
  __bf16* ff1   = (__bf16*)alloc((size_t)NTOK * DFF_ * 2);

  // --- precision prep: fp32 -> bf16 (weights transposed to [N][K]) --------
  cast_bf16_kernel<<<(NTOK * D_) / 256, 256, 0, stream>>>(x, xb, NTOK * D_);
  transpose_cast_kernel<<<(D_ * D_) / 256, 256, 0, stream>>>(Wq, wqkvT, D_, D_);
  transpose_cast_kernel<<<(D_ * D_) / 256, 256, 0, stream>>>(Wk, wqkvT + D_ * D_, D_, D_);
  transpose_cast_kernel<<<(D_ * D_) / 256, 256, 0, stream>>>(Wv, wqkvT + 2 * D_ * D_, D_, D_);
  transpose_cast_kernel<<<(D_ * D_) / 256, 256, 0, stream>>>(Wo, woT, D_, D_);
  transpose_cast_kernel<<<(D_ * DFF_) / 256, 256, 0, stream>>>(W1, w1T, D_, DFF_);
  transpose_cast_kernel<<<(D_ * DFF_) / 256, 256, 0, stream>>>(W2, w2T, DFF_, D_);
  concat_bias_kernel<<<12, 256, 0, stream>>>(bq, bk, bv, bqkv);

  // --- fused QKV projection: [8192,1024] @ [1024,3072] -> bf16 ------------
  gemm_wmma<false, true><<<dim3(3 * D_ / 128, NTOK / 128), 256, 0, stream>>>(
      xb, wqkvT, bqkv, nullptr, qkv, NTOK, 3 * D_, D_);

  // --- per-head V transpose for async-streamable B-fragments --------------
  vtrans_kernel<<<dim3(T_ / 64, B_ * H_), 256, 0, stream>>>(qkv, vt);

  // --- flash attention with relative-position bias ------------------------
  attn_wmma<<<dim3(T_ / 128, B_ * H_), 256, 0, stream>>>(qkv, vt, bias_table, ctxb);

  // --- output projection ---------------------------------------------------
  gemm_wmma<false, false><<<dim3(D_ / 128, NTOK / 128), 256, 0, stream>>>(
      ctxb, woT, bo, attnf, nullptr, NTOK, D_, D_);

  // --- x1 = LN(x + attn_out) ----------------------------------------------
  add_ln<<<NTOK, 256, 0, stream>>>(x, attnf, ln1_g, ln1_b, x1f, x1b);

  // --- FFN: relu(x1 @ W1 + b1) @ W2 + b2 ----------------------------------
  gemm_wmma<true, true><<<dim3(DFF_ / 128, NTOK / 128), 256, 0, stream>>>(
      x1b, w1T, b1, nullptr, ff1, NTOK, DFF_, D_);
  gemm_wmma<false, false><<<dim3(D_ / 128, NTOK / 128), 256, 0, stream>>>(
      ff1, w2T, b2, attnf, nullptr, NTOK, D_, DFF_);

  // --- out = LN(x1 + ff) ---------------------------------------------------
  add_ln<<<NTOK, 256, 0, stream>>>(x1f, attnf, ln2_g, ln2_b, out, nullptr);
}
